// Local_Inference_Modeling_39041252721267
// MI455X (gfx1250) — matrix operand order repaired and checked by reading, not compiled, not run on hardware
//
#include <hip/hip_runtime.h>
#include <hip/hip_bf16.h>

// ---------------------------------------------------------------------------
// MI455X (gfx1250) fused bidirectional-attention (ESIM-style) pipeline.
// 5 kernels: e-GEMM (WMMA bf16), row-softmax, col-softmax, a-side GEMM+epilogue,
// b-side GEMM+epilogue. All matmuls via v_wmma_f32_16x16x32_bf16, f32 accum.
// Workspace layout (48 MB, L2-resident):
//   [0,   25165824)  e        fp32  [32][512][384]
//   [+0,  12582912)  att_a    bf16  [32][512][384]  (row softmax)
//   [+.., 12582912)  att_b    bf16  [32][512][384]  (col softmax, same layout)
// ---------------------------------------------------------------------------

#define BATCH 32
#define LA    512
#define LB    384
#define DD    1024
#define LDSS  40      // LDS row stride in bf16 elements (80B: 16B-aligned, bank-friendly)

typedef __attribute__((ext_vector_type(16))) __bf16        v16bf;
typedef __attribute__((ext_vector_type(8)))  float         v8f;
typedef __attribute__((ext_vector_type(8)))  unsigned short v8s;

__device__ __forceinline__ unsigned short f2bf(float f) {
  union { float f; unsigned u; } x; x.f = f;
  unsigned r = x.u + 0x7FFFu + ((x.u >> 16) & 1u);   // round-to-nearest-even
  return (unsigned short)(r >> 16);
}

// ---- LDS staging (blockDim == 256) ----------------------------------------
// Direct: global fp32 [128 rows x 32 k] -> LDS bf16 [128][LDSS]
__device__ __forceinline__ void stage_d_f32(unsigned short* dst, const float* src, int ldg) {
  int t  = threadIdx.x;
  int r  = t >> 1;             // 0..127
  int ks = (t & 1) << 4;       // 0 | 16
  const float4* p = (const float4*)(src + (size_t)r * ldg + ks);
  float4 q0 = p[0], q1 = p[1], q2 = p[2], q3 = p[3];
  float f[16] = {q0.x,q0.y,q0.z,q0.w, q1.x,q1.y,q1.z,q1.w,
                 q2.x,q2.y,q2.z,q2.w, q3.x,q3.y,q3.z,q3.w};
  union { v8s v[2]; unsigned short h[16]; } u;
#pragma unroll
  for (int l = 0; l < 16; ++l) u.h[l] = f2bf(f[l]);
  v8s* d = (v8s*)(dst + r * LDSS + ks);
  d[0] = u.v[0]; d[1] = u.v[1];
}

// Direct: global bf16 [128 rows x 32 k] -> LDS bf16 [128][LDSS]
__device__ __forceinline__ void stage_d_bf16(unsigned short* dst, const unsigned short* src, int ldg) {
  int t  = threadIdx.x;
  int r  = t >> 1;
  int ks = (t & 1) << 4;
  const v8s* p = (const v8s*)(src + (size_t)r * ldg + ks);
  v8s a0 = p[0], a1 = p[1];
  v8s* d = (v8s*)(dst + r * LDSS + ks);
  d[0] = a0; d[1] = a1;
}

// Transposed: global fp32 [32 k-rows x 128 cols] -> LDS bf16 [col 128][k 32]
__device__ __forceinline__ void stage_t_f32(unsigned short* dst, const float* src, int ldg) {
  int t  = threadIdx.x;
  int kr = t >> 3;             // 0..31
  int cs = (t & 7) << 4;       // 0..112
  const float4* p = (const float4*)(src + (size_t)kr * ldg + cs);
  float4 q0 = p[0], q1 = p[1], q2 = p[2], q3 = p[3];
  float f[16] = {q0.x,q0.y,q0.z,q0.w, q1.x,q1.y,q1.z,q1.w,
                 q2.x,q2.y,q2.z,q2.w, q3.x,q3.y,q3.z,q3.w};
#pragma unroll
  for (int l = 0; l < 16; ++l) dst[(cs + l) * LDSS + kr] = f2bf(f[l]);
}

// Transposed: global bf16 [32 k-rows x 128 cols] -> LDS bf16 [col 128][k 32]
__device__ __forceinline__ void stage_t_bf16(unsigned short* dst, const unsigned short* src, int ldg) {
  int t  = threadIdx.x;
  int kr = t >> 3;
  int cs = (t & 7) << 4;
  const v8s* p = (const v8s*)(src + (size_t)kr * ldg + cs);
  union { v8s v[2]; unsigned short h[16]; } u;
  u.v[0] = p[0]; u.v[1] = p[1];
#pragma unroll
  for (int l = 0; l < 16; ++l) dst[(cs + l) * LDSS + kr] = u.h[l];
}

// ---- WMMA fragment loads (wave32, layouts per CDNA5 ISA 7.12.2) -----------
// A 16x32 bf16: lane l holds row l%16; K = (l/16)*8 + {0..7} and 16+(l/16)*8 + {0..7}
__device__ __forceinline__ v16bf frag_a(const unsigned short* lds, int m0) {
  int l   = threadIdx.x & 31;
  int row = m0 + (l & 15);
  int kb  = (l >> 4) << 3;
  const unsigned short* p = lds + row * LDSS + kb;
  union { v8s h[2]; v16bf v; } u;
  u.h[0] = *(const v8s*)p;
  u.h[1] = *(const v8s*)(p + 16);
  return u.v;
}
// B 32x16 bf16 (from Bt[n][k] in LDS): lane l holds col l%16; K = (l/16)*8 + {0..7} and 16+(l/16)*8 + {0..7}
__device__ __forceinline__ v16bf frag_b(const unsigned short* lds, int n0) {
  int l   = threadIdx.x & 31;
  int row = n0 + (l & 15);
  int kh  = (l >> 4) << 3;
  const unsigned short* p = lds + row * LDSS + kh;
  union { v8s h[2]; v16bf v; } u;
  u.h[0] = *(const v8s*)p;
  u.h[1] = *(const v8s*)(p + 16);
  return u.v;
}

// One K=32 step: wave computes 32x64 (2x4 tiles of 16x16)
__device__ __forceinline__ void mma_step(const unsigned short* As, const unsigned short* Bs,
                                         int wm, int wn, v8f acc[2][4]) {
  v16bf af[2], bf[4];
#pragma unroll
  for (int tm = 0; tm < 2; ++tm) af[tm] = frag_a(As, wm * 32 + tm * 16);
#pragma unroll
  for (int tn = 0; tn < 4; ++tn) bf[tn] = frag_b(Bs, wn * 64 + tn * 16);
#pragma unroll
  for (int tm = 0; tm < 2; ++tm)
#pragma unroll
    for (int tn = 0; tn < 4; ++tn)
      acc[tm][tn] = __builtin_amdgcn_wmma_f32_16x16x32_bf16(
          false, af[tm], false, bf[tn], (short)0, acc[tm][tn], false, false);
}

// ---------------------------------------------------------------------------
// Kernel 1: e[b,i,j] = sum_d A[b,i,d]*B[b,j,d].  Block = 128x128, K=1024.
__global__ __launch_bounds__(256) void k_gemm_e(const float* __restrict__ A,
                                                const float* __restrict__ B,
                                                float* __restrict__ E) {
  int bx = blockIdx.x;
  int jblk = bx % 3, iblk = (bx / 3) % 4, b = bx / 12;
  const float* Ab = A + ((size_t)b * LA + (size_t)iblk * 128) * DD;
  const float* Bb = B + ((size_t)b * LB + (size_t)jblk * 128) * DD;
  float*       Eb = E + (size_t)b * LA * LB + (size_t)iblk * 128 * LB + jblk * 128;

  __shared__ unsigned short As[2][128 * LDSS];
  __shared__ unsigned short Bs[2][128 * LDSS];
  int w = threadIdx.x >> 5, wm = w & 3, wn = w >> 2;
  v8f zero = {0.f,0.f,0.f,0.f,0.f,0.f,0.f,0.f};
  v8f acc[2][4];
#pragma unroll
  for (int tm = 0; tm < 2; ++tm)
#pragma unroll
    for (int tn = 0; tn < 4; ++tn) acc[tm][tn] = zero;

  stage_d_f32(As[0], Ab, DD);
  stage_d_f32(Bs[0], Bb, DD);
  __syncthreads();
  int buf = 0;
  for (int ks = 0; ks < 32; ++ks) {
    if (ks + 1 < 32) {
      stage_d_f32(As[buf ^ 1], Ab + (ks + 1) * 32, DD);
      stage_d_f32(Bs[buf ^ 1], Bb + (ks + 1) * 32, DD);
    }
    mma_step(As[buf], Bs[buf], wm, wn, acc);
    __syncthreads();
    buf ^= 1;
  }
  int lane = threadIdx.x & 31;
  int nloc = lane & 15, rh = (lane >> 4) * 8;
#pragma unroll
  for (int tm = 0; tm < 2; ++tm)
#pragma unroll
    for (int tn = 0; tn < 4; ++tn)
#pragma unroll
      for (int r = 0; r < 8; ++r) {
        int i = wm * 32 + tm * 16 + rh + r;
        int j = wn * 64 + tn * 16 + nloc;
        Eb[(size_t)i * LB + j] = acc[tm][tn][r];
      }
}

// Kernel 2a: row softmax over j (one wave per row), att_a bf16 [b][i][j]
__global__ __launch_bounds__(256) void k_soft_row(const float* __restrict__ E,
                                                  unsigned short* __restrict__ ATT) {
  int w = threadIdx.x >> 5, lane = threadIdx.x & 31;
  int row = blockIdx.x * 8 + w;                     // 0..16383
  const float* e = E + (size_t)row * LB;
  float v[12];
  float m = -1e30f;
#pragma unroll
  for (int t = 0; t < 12; ++t) { v[t] = e[lane + 32 * t]; m = fmaxf(m, v[t]); }
#pragma unroll
  for (int o = 16; o; o >>= 1) m = fmaxf(m, __shfl_xor(m, o, 32));
  float s = 0.f;
#pragma unroll
  for (int t = 0; t < 12; ++t) { v[t] = __expf(v[t] - m); s += v[t]; }
#pragma unroll
  for (int o = 16; o; o >>= 1) s += __shfl_xor(s, o, 32);
  float inv = 1.0f / s;
  unsigned short* a = ATT + (size_t)row * LB;
#pragma unroll
  for (int t = 0; t < 12; ++t) a[lane + 32 * t] = f2bf(v[t] * inv);
}

// Kernel 2b: col softmax over i (one thread per column), att_b bf16 [b][i][j]
__global__ __launch_bounds__(256) void k_soft_col(const float* __restrict__ E,
                                                  unsigned short* __restrict__ ATTB) {
  int t = blockIdx.x * 256 + threadIdx.x;           // 0..12287
  int b = t / LB, j = t % LB;
  const float* e = E + (size_t)b * LA * LB + j;
  float m = -1e30f, s = 0.f;
  for (int i = 0; i < LA; ++i) {                    // online max+sum
    float x = e[(size_t)i * LB];
    float nm = fmaxf(m, x);
    s = s * __expf(m - nm) + __expf(x - nm);
    m = nm;
  }
  float inv = 1.0f / s;
  unsigned short* o = ATTB + (size_t)b * LA * LB + j;
  for (int i = 0; i < LA; ++i) {
    float x = e[(size_t)i * LB];
    o[(size_t)i * LB] = f2bf(__expf(x - m) * inv);
  }
}

// Kernel 3: a_tilde = att_a @ B, fused m_a epilogue.  M=i(512), K=j(384), N=d(1024)
__global__ __launch_bounds__(256) void k_out_a(const unsigned short* __restrict__ ATT,
                                               const float* __restrict__ Bbar,
                                               const float* __restrict__ Abar,
                                               float* __restrict__ MA) {
  int bx = blockIdx.x;
  int dblk = bx & 7, iblk = (bx >> 3) & 3, b = bx >> 5;
  const unsigned short* att = ATT + (size_t)b * LA * LB + (size_t)iblk * 128 * LB;
  const float* Bb = Bbar + (size_t)b * LB * DD + dblk * 128;

  __shared__ unsigned short As[2][128 * LDSS];
  __shared__ unsigned short Bs[2][128 * LDSS];
  int w = threadIdx.x >> 5, wm = w & 3, wn = w >> 2;
  v8f zero = {0.f,0.f,0.f,0.f,0.f,0.f,0.f,0.f};
  v8f acc[2][4];
#pragma unroll
  for (int tm = 0; tm < 2; ++tm)
#pragma unroll
    for (int tn = 0; tn < 4; ++tn) acc[tm][tn] = zero;

  stage_d_bf16(As[0], att, LB);
  stage_t_f32 (Bs[0], Bb, DD);
  __syncthreads();
  int buf = 0;
  for (int ks = 0; ks < 12; ++ks) {                 // K = 384
    if (ks + 1 < 12) {
      stage_d_bf16(As[buf ^ 1], att + (ks + 1) * 32, LB);
      stage_t_f32 (Bs[buf ^ 1], Bb + (size_t)(ks + 1) * 32 * DD, DD);
    }
    mma_step(As[buf], Bs[buf], wm, wn, acc);
    __syncthreads();
    buf ^= 1;
  }
  const float* Ab = Abar + ((size_t)b * LA + (size_t)iblk * 128) * DD + dblk * 128;
  float* MAb = MA + ((size_t)b * LA + (size_t)iblk * 128) * 4096 + dblk * 128;
  int lane = threadIdx.x & 31;
  int nloc = lane & 15, rh = (lane >> 4) * 8;
#pragma unroll
  for (int tm = 0; tm < 2; ++tm)
#pragma unroll
    for (int tn = 0; tn < 4; ++tn)
#pragma unroll
      for (int r = 0; r < 8; ++r) {
        int i = wm * 32 + tm * 16 + rh + r;
        int d = wn * 64 + tn * 16 + nloc;
        float at = acc[tm][tn][r];
        float av = Ab[(size_t)i * DD + d];
        float* o = MAb + (size_t)i * 4096 + d;
        o[0] = av; o[1024] = at; o[2048] = av - at; o[3072] = av * at;
      }
}

// Kernel 4: b_tilde = att_bᵀ @ A, fused m_b epilogue.  M=j(384), K=i(512), N=d(1024)
__global__ __launch_bounds__(256) void k_out_b(const unsigned short* __restrict__ ATTB,
                                               const float* __restrict__ Abar,
                                               const float* __restrict__ Bbar,
                                               float* __restrict__ MB) {
  int bx = blockIdx.x;
  int dblk = bx & 7, jblk = (bx >> 3) % 3, b = bx / 24;
  const unsigned short* attb = ATTB + (size_t)b * LA * LB + jblk * 128;   // [i][j] layout
  const float* Ab = Abar + (size_t)b * LA * DD + dblk * 128;

  __shared__ unsigned short As[2][128 * LDSS];
  __shared__ unsigned short Bs[2][128 * LDSS];
  int w = threadIdx.x >> 5, wm = w & 3, wn = w >> 2;
  v8f zero = {0.f,0.f,0.f,0.f,0.f,0.f,0.f,0.f};
  v8f acc[2][4];
#pragma unroll
  for (int tm = 0; tm < 2; ++tm)
#pragma unroll
    for (int tn = 0; tn < 4; ++tn) acc[tm][tn] = zero;

  stage_t_bf16(As[0], attb, LB);                    // -> LDS [j 128][i 32]
  stage_t_f32 (Bs[0], Ab, DD);                      // -> LDS [d 128][i 32]
  __syncthreads();
  int buf = 0;
  for (int ks = 0; ks < 16; ++ks) {                 // K = 512
    if (ks + 1 < 16) {
      stage_t_bf16(As[buf ^ 1], attb + (size_t)(ks + 1) * 32 * LB, LB);
      stage_t_f32 (Bs[buf ^ 1], Ab + (size_t)(ks + 1) * 32 * DD, DD);
    }
    mma_step(As[buf], Bs[buf], wm, wn, acc);
    __syncthreads();
    buf ^= 1;
  }
  const float* Bb = Bbar + ((size_t)b * LB + (size_t)jblk * 128) * DD + dblk * 128;
  float* MBb = MB + ((size_t)b * LB + (size_t)jblk * 128) * 4096 + dblk * 128;
  int lane = threadIdx.x & 31;
  int nloc = lane & 15, rh = (lane >> 4) * 8;
#pragma unroll
  for (int tm = 0; tm < 2; ++tm)
#pragma unroll
    for (int tn = 0; tn < 4; ++tn)
#pragma unroll
      for (int r = 0; r < 8; ++r) {
        int j = wm * 32 + tm * 16 + rh + r;
        int d = wn * 64 + tn * 16 + nloc;
        float bt = acc[tm][tn][r];
        float bv = Bb[(size_t)j * DD + d];
        float* o = MBb + (size_t)j * 4096 + d;
        o[0] = bv; o[1024] = bt; o[2048] = bv - bt; o[3072] = bv * bt;
      }
}

// ---------------------------------------------------------------------------
extern "C" void kernel_launch(void* const* d_in, const int* in_sizes, int n_in,
                              void* d_out, int out_size, void* d_ws, size_t ws_size,
                              hipStream_t stream) {
  (void)in_sizes; (void)n_in; (void)out_size; (void)ws_size;
  const float* a_bar = (const float*)d_in[0];   // [32,512,1024] fp32
  const float* b_bar = (const float*)d_in[1];   // [32,384,1024] fp32
  float* out = (float*)d_out;
  float* MA  = out;                                        // [32,512,4096]
  float* MB  = out + (size_t)BATCH * LA * 4096;            // [32,384,4096]

  char* ws = (char*)d_ws;                                  // needs 48 MB
  float*          E    = (float*)ws;                                    // 25165824 B
  unsigned short* ATTA = (unsigned short*)(ws + 25165824);              // 12582912 B
  unsigned short* ATTB = (unsigned short*)(ws + 25165824 + 12582912);   // 12582912 B

  k_gemm_e <<<BATCH * 4 * 3, 256, 0, stream>>>(a_bar, b_bar, E);
  k_soft_row<<<(BATCH * LA) / 8, 256, 0, stream>>>(E, ATTA);
  k_soft_col<<<(BATCH * LB) / 256, 256, 0, stream>>>(E, ATTB);
  k_out_a  <<<BATCH * 4 * 8, 256, 0, stream>>>(ATTA, b_bar, a_bar, MA);
  k_out_b  <<<BATCH * 3 * 8, 256, 0, stream>>>(ATTB, a_bar, b_bar, MB);
}
